// GroupedQueryAttention_83073257439622
// MI455X (gfx1250) — compile-verified, matrix operand order checked by reading, not run on hardware
//
#include <hip/hip_runtime.h>
#include <hip/hip_fp16.h>

typedef __attribute__((ext_vector_type(16))) _Float16 v16h;
typedef __attribute__((ext_vector_type(8)))  float    v8f;

#define SEQ   1024
#define DIN   4096
#define HEADS 32
#define GRPS  8
#define HD    128

#define WMMA16(a, b, c) \
  __builtin_amdgcn_wmma_f32_16x16x32_f16(false, (a), false, (b), (short)0, (c), false, false)

__device__ inline unsigned short f2h_bits(float f) {
  _Float16 h = (_Float16)f;
  union { _Float16 h; unsigned short u; } c; c.h = h; return c.u;
}
__device__ inline unsigned int pack2(float a, float b) {
  return (unsigned int)f2h_bits(a) | ((unsigned int)f2h_bits(b) << 16);
}
__device__ inline v8f zero8() {
  v8f z;
#pragma unroll
  for (int i = 0; i < 8; ++i) z[i] = 0.f;
  return z;
}

// Load a 16x32 f16 fragment in the CDNA5 WMMA 16-bit A layout from LDS.
// lanes 0-15: row = lane, K pairs {0..7} in v0-3, {16..23} in v4-7
// lanes 16-31: row = lane-16, K pairs {8..15} in v0-3, {24..31} in v4-7
__device__ inline v16h ldfrag(const unsigned short* t, int stride, int row0,
                              int kofs, int lane) {
  int r  = row0 + (lane & 15);
  int kb = kofs + ((lane & 16) ? 8 : 0);
  const unsigned short* p = t + r * stride + kb;
  union { uint4 q[2]; v16h v; } u;
  u.q[0] = *(const uint4*)(p);
  u.q[1] = *(const uint4*)(p + 16);
  return u.v;
}

// C(1024xN) = A(1024x4096 f32) x B(4096xN f32), fp32 accum via WMMA f16.
// Block 256 threads = 8 waves; 128x128 tile; wave tile 64x32 (4x2 frags).
// Software-pipelined: next k-tile's global loads overlap current WMMAs.
template <int N>
__global__ __launch_bounds__(256) void gemm_wmma(const float* __restrict__ A,
                                                 const float* __restrict__ B,
                                                 float* __restrict__ C) {
  constexpr int K = 4096;
  __shared__ unsigned short Ah[128][40];   // [m][k] f16, padded
  __shared__ unsigned short Bt[128][40];   // [n][k] f16 (B transposed), padded
  const int tid = threadIdx.x, lane = tid & 31, wv = tid >> 5;
  const int wr = wv >> 2, wc = wv & 3;
  const int m0 = blockIdx.y * 128, n0 = blockIdx.x * 128;

  // Per-thread fill coordinates (invariant across the 4 unrolled chunks).
  const int ar0 = tid >> 3, ac = (tid & 7) << 2;    // A: rows ar0+32i, col ac
  const int br0 = tid >> 5, bc = (tid & 31) << 2;   // B: rows br0+8i,  col bc

  v8f acc[4][2];
#pragma unroll
  for (int i = 0; i < 4; ++i)
#pragma unroll
    for (int j = 0; j < 2; ++j) acc[i][j] = zero8();

  float4 pa[4], pb[4];
  // prologue: fetch k-tile 0
#pragma unroll
  for (int i = 0; i < 4; ++i) {
    pa[i] = *(const float4*)(A + (size_t)(m0 + ar0 + 32 * i) * K + ac);
    pb[i] = *(const float4*)(B + (size_t)(br0 + 8 * i) * N + n0 + bc);
  }

  for (int kt = 0; kt < K; kt += 32) {
    // commit prefetched tile to LDS (converted to f16)
#pragma unroll
    for (int i = 0; i < 4; ++i) {
      uint2 ap; ap.x = pack2(pa[i].x, pa[i].y); ap.y = pack2(pa[i].z, pa[i].w);
      *(uint2*)&Ah[ar0 + 32 * i][ac] = ap;
      Bt[bc + 0][br0 + 8 * i] = f2h_bits(pb[i].x);
      Bt[bc + 1][br0 + 8 * i] = f2h_bits(pb[i].y);
      Bt[bc + 2][br0 + 8 * i] = f2h_bits(pb[i].z);
      Bt[bc + 3][br0 + 8 * i] = f2h_bits(pb[i].w);
    }
    __syncthreads();
    // prefetch next k-tile while WMMAs below execute
    if (kt + 32 < K) {
#pragma unroll
      for (int i = 0; i < 4; ++i) {
        pa[i] = *(const float4*)(A + (size_t)(m0 + ar0 + 32 * i) * K + (kt + 32) + ac);
        pb[i] = *(const float4*)(B + (size_t)(kt + 32 + br0 + 8 * i) * N + n0 + bc);
      }
    }
    v16h bf0 = ldfrag(&Bt[0][0], 40, 32 * wc,      0, lane);
    v16h bf1 = ldfrag(&Bt[0][0], 40, 32 * wc + 16, 0, lane);
#pragma unroll
    for (int fm = 0; fm < 4; ++fm) {
      v16h af = ldfrag(&Ah[0][0], 40, 64 * wr + 16 * fm, 0, lane);
      acc[fm][0] = WMMA16(af, bf0, acc[fm][0]);
      acc[fm][1] = WMMA16(af, bf1, acc[fm][1]);
    }
    __syncthreads();
  }
  const int hi = (lane & 16) ? 8 : 0;
#pragma unroll
  for (int fm = 0; fm < 4; ++fm) {
    const int mb = m0 + 64 * wr + 16 * fm + hi;
#pragma unroll
    for (int fn = 0; fn < 2; ++fn) {
      float* cp = C + (size_t)mb * N + n0 + 32 * wc + 16 * fn + (lane & 15);
#pragma unroll
      for (int r = 0; r < 8; ++r)
        cp[r * N] = acc[fm][fn][r];     // compile-time immediate offsets
    }
  }
}

__global__ void rope_f16(const float* __restrict__ in, const float* __restrict__ cs,
                         const float* __restrict__ sn, _Float16* __restrict__ out,
                         int width) {
  int idx = blockIdx.x * blockDim.x + threadIdx.x;
  int s = idx / width;
  int c = idx - s * width;
  int d = c & (HD - 1);
  float v   = in[idx];
  float rot = (d < HD / 2) ? -in[idx + HD / 2] : in[idx - HD / 2];
  out[idx] = (_Float16)(v * cs[s * HD + d] + rot * sn[s * HD + d]);
}

__global__ void cvt_f16(const float* __restrict__ in, _Float16* __restrict__ out) {
  int idx = blockIdx.x * blockDim.x + threadIdx.x;
  out[idx] = (_Float16)in[idx];
}

// Flash-style causal GQA. Block = 64 queries x 1 head, 128 threads (4 waves).
// Wave w owns query rows sq..sq+15; iterates 32-key tiles shared via LDS.
__global__ __launch_bounds__(128) void attn_wmma(const _Float16* __restrict__ Q,
                                                 const _Float16* __restrict__ Kc,
                                                 const _Float16* __restrict__ Vc,
                                                 float* __restrict__ ctx) {
  __shared__ unsigned short Kl[32][136];   // [key][d]  f16
  __shared__ unsigned short Vt[128][40];   // [d][key]  f16 (transposed)
  __shared__ unsigned short Pl[4][16][40]; // per-wave P relayout buffer
  const int tid = threadIdx.x, lane = tid & 31, w = tid >> 5;
  const int h = blockIdx.y, g = h >> 2;           // GS = H/G = 4
  const int s0 = blockIdx.x * 64, sq = s0 + 16 * w;
  const int hi16 = (lane & 16) ? 8 : 0;

  // Q fragments for all 4 k-steps (d = 0..127), loaded once from global.
  v16h qf[4];
  {
    const _Float16* qb =
        Q + (size_t)(sq + (lane & 15)) * (HEADS * HD) + h * HD + hi16;
#pragma unroll
    for (int ks = 0; ks < 4; ++ks) {
      union { uint4 q[2]; v16h v; } u;
      u.q[0] = *(const uint4*)(qb + ks * 32);
      u.q[1] = *(const uint4*)(qb + ks * 32 + 16);
      qf[ks] = u.v;
    }
  }

  v8f o[8];
  float mi[8], li[8];
#pragma unroll
  for (int i = 0; i < 8; ++i) { o[i] = zero8(); mi[i] = -__builtin_inff(); li[i] = 0.f; }
  const float scale = 0.08838834764831845f;       // 1/sqrt(128)
  const int ntiles = (s0 + 64) >> 5;              // keys needed: t <= s0+63

  for (int t = 0; t < ntiles; ++t) {
    const int t0 = t << 5;
    // Cooperative stage of K (as-is) and V (transposed) tiles.
#pragma unroll
    for (int i = 0; i < 4; ++i) {
      int idx = tid + 128 * i;                     // 0..511
      int kr = idx >> 4, c0 = (idx & 15) << 3;
      const size_t rb = (size_t)(t0 + kr) * (GRPS * HD) + g * HD + c0;
      *(uint4*)&Kl[kr][c0] = *(const uint4*)(Kc + rb);
      union { uint4 q; unsigned short s[8]; } uv;
      uv.q = *(const uint4*)(Vc + rb);
#pragma unroll
      for (int j = 0; j < 8; ++j) Vt[c0 + j][kr] = uv.s[j];
    }
    __syncthreads();

    // S = Q @ K^T  (16 x 32 scores, two 16x16 fragments)
    v8f sf0 = zero8(), sf1 = zero8();
#pragma unroll
    for (int ks = 0; ks < 4; ++ks) {
      v16h b0 = ldfrag(&Kl[0][0], 136, 0,  ks * 32, lane);
      v16h b1 = ldfrag(&Kl[0][0], 136, 16, ks * 32, lane);
      sf0 = WMMA16(qf[ks], b0, sf0);
      sf1 = WMMA16(qf[ks], b1, sf1);
    }

    // Causal mask + scale + online softmax (row stats per lane half).
    const int tc0 = t0 + (lane & 15);
    float pr0[8], pr1[8];
#pragma unroll
    for (int r = 0; r < 8; ++r) {
      int srow = sq + hi16 + r;
      float x0 = (tc0      > srow) ? -__builtin_inff() : sf0[r] * scale;
      float x1 = (tc0 + 16 > srow) ? -__builtin_inff() : sf1[r] * scale;
      float rm = fmaxf(x0, x1);
      rm = fmaxf(rm, __shfl_xor(rm, 1, 32));
      rm = fmaxf(rm, __shfl_xor(rm, 2, 32));
      rm = fmaxf(rm, __shfl_xor(rm, 4, 32));
      rm = fmaxf(rm, __shfl_xor(rm, 8, 32));
      float mn = fmaxf(mi[r], rm);
      float alpha = __expf(mi[r] - mn);
      float p0 = __expf(x0 - mn);
      float p1 = __expf(x1 - mn);
      float rs = p0 + p1;
      rs += __shfl_xor(rs, 1, 32);
      rs += __shfl_xor(rs, 2, 32);
      rs += __shfl_xor(rs, 4, 32);
      rs += __shfl_xor(rs, 8, 32);
      li[r] = li[r] * alpha + rs;
      mi[r] = mn;
#pragma unroll
      for (int j = 0; j < 8; ++j) o[j][r] *= alpha;
      pr0[r] = p0; pr1[r] = p1;
    }

    // Re-layout P: C-format -> LDS -> A-format, then O += P @ V.
#pragma unroll
    for (int r = 0; r < 8; ++r) {
      Pl[w][hi16 + r][lane & 15]        = f2h_bits(pr0[r]);
      Pl[w][hi16 + r][16 + (lane & 15)] = f2h_bits(pr1[r]);
    }
    __syncthreads();
    v16h pa = ldfrag(&Pl[w][0][0], 40, 0, 0, lane);
#pragma unroll
    for (int fn = 0; fn < 8; ++fn) {
      v16h vb = ldfrag(&Vt[0][0], 40, 16 * fn, 0, lane);
      o[fn] = WMMA16(pa, vb, o[fn]);
    }
    __syncthreads();
  }

  float invl[8];
#pragma unroll
  for (int r = 0; r < 8; ++r) invl[r] = 1.0f / li[r];
#pragma unroll
  for (int fn = 0; fn < 8; ++fn) {
    float* cp = ctx + (size_t)(sq + hi16) * (HEADS * HD) + h * HD + 16 * fn + (lane & 15);
#pragma unroll
    for (int r = 0; r < 8; ++r)
      cp[r * (HEADS * HD)] = o[fn][r] * invl[r];   // immediate offsets
  }
}

extern "C" void kernel_launch(void* const* d_in, const int* in_sizes, int n_in,
                              void* d_out, int out_size, void* d_ws, size_t ws_size,
                              hipStream_t stream) {
  (void)in_sizes; (void)n_in; (void)out_size; (void)ws_size;
  const float* x  = (const float*)d_in[0];
  // d_in[1] = mask (unused; causal mask computed analytically)
  const float* cs = (const float*)d_in[2];
  const float* sn = (const float*)d_in[3];
  const float* Wq = (const float*)d_in[4];
  const float* Wk = (const float*)d_in[5];
  const float* Wv = (const float*)d_in[6];
  const float* Wo = (const float*)d_in[7];
  float* out = (float*)d_out;

  char* ws = (char*)d_ws;
  float* qf32 = (float*)ws;       ws += (size_t)SEQ * DIN * 4;
  float* kf32 = (float*)ws;       ws += (size_t)SEQ * GRPS * HD * 4;
  float* vf32 = (float*)ws;       ws += (size_t)SEQ * GRPS * HD * 4;
  float* ctx  = (float*)ws;       ws += (size_t)SEQ * DIN * 4;
  _Float16* qh = (_Float16*)ws;   ws += (size_t)SEQ * DIN * 2;
  _Float16* kh = (_Float16*)ws;   ws += (size_t)SEQ * GRPS * HD * 2;
  _Float16* vh = (_Float16*)ws;

  gemm_wmma<DIN><<<dim3(DIN / 128, SEQ / 128), 256, 0, stream>>>(x, Wq, qf32);
  gemm_wmma<GRPS * HD><<<dim3((GRPS * HD) / 128, SEQ / 128), 256, 0, stream>>>(x, Wk, kf32);
  gemm_wmma<GRPS * HD><<<dim3((GRPS * HD) / 128, SEQ / 128), 256, 0, stream>>>(x, Wv, vf32);
  rope_f16<<<(SEQ * DIN) / 256, 256, 0, stream>>>(qf32, cs, sn, qh, DIN);
  rope_f16<<<(SEQ * GRPS * HD) / 256, 256, 0, stream>>>(kf32, cs, sn, kh, GRPS * HD);
  cvt_f16<<<(SEQ * GRPS * HD) / 256, 256, 0, stream>>>(vf32, vh);
  attn_wmma<<<dim3(SEQ / 64, HEADS), 128, 0, stream>>>(qh, kh, vh, ctx);
  gemm_wmma<DIN><<<dim3(DIN / 128, SEQ / 128), 256, 0, stream>>>(ctx, Wo, out);
}
